// GraphEncoder_66589172957276
// MI455X (gfx1250) — matrix operand order repaired and checked by reading, not compiled, not run on hardware
//
#include <hip/hip_runtime.h>
#include <stdint.h>

typedef __attribute__((ext_vector_type(16))) __bf16 bf16x16;
typedef __attribute__((ext_vector_type(8)))  float  v8f;
typedef __attribute__((ext_vector_type(4)))  unsigned int u32x4;
typedef __attribute__((ext_vector_type(8)))  int  i32x8;
typedef __attribute__((ext_vector_type(4)))  int  i32x4;

#define OUTD 128

#if __has_builtin(__builtin_amdgcn_tensor_load_to_lds)
#define HAVE_TDM 1
#if __has_include(<hip/amd_detail/amd_gfx1250_TDM.h>)
// therock-10.0 headers -> clang-23 6-arg builtin
#define TDM_LOAD(g0, g1, g2, g3) \
  __builtin_amdgcn_tensor_load_to_lds((g0), (g1), (g2), (g3), (i32x8){0,0,0,0,0,0,0,0}, 0)
#else
// ROCm 7.2 clang-22 5-arg builtin
#define TDM_LOAD(g0, g1, g2, g3) \
  __builtin_amdgcn_tensor_load_to_lds((g0), (g1), (g2), (g3), 0)
#endif
#endif

__device__ __forceinline__ unsigned short f32_to_bf16_bits(float f) {
  union { float f; unsigned u; } c; c.f = f;
  unsigned u = c.u;
  u += 0x7FFFu + ((u >> 16) & 1u);   // round-to-nearest-even
  return (unsigned short)(u >> 16);
}
__device__ __forceinline__ __bf16 f32_to_bf16(float f) {
  union { unsigned short s; __bf16 b; } c;
  c.s = f32_to_bf16_bits(f);
  return c.b;
}

// Native L2 float atomic (no-return form, STOREcnt-tracked; S_ENDPGM implies wait-idle).
// Avoids any risk of AtomicExpand CAS-loop lowering on the hottest path.
__device__ __forceinline__ void atomic_add_f32(float* p, float v) {
  asm volatile("global_atomic_add_f32 %0, %1, off" :: "v"(p), "v"(v) : "memory");
}

// ---------------- small elementwise kernels ----------------

__global__ __launch_bounds__(256) void k_fill(float* __restrict__ p, float v, int n) {
  int i = blockIdx.x * 256 + threadIdx.x;
  if (i < n) p[i] = v;
}

__global__ __launch_bounds__(256) void k_deg_accum(const long long* __restrict__ ei,
                                                   const float* __restrict__ ew,
                                                   float* __restrict__ deg, int E) {
  int e = blockIdx.x * 256 + threadIdx.x;
  if (e < E) {
    int c = (int)ei[(size_t)E + e];
    atomic_add_f32(&deg[c], ew[e]);
  }
}

__global__ __launch_bounds__(256) void k_rsqrt(float* __restrict__ p, int n) {
  int i = blockIdx.x * 256 + threadIdx.x;
  if (i < n) p[i] = rsqrtf(p[i]);   // deg >= 2.0 always (self loop), no zero guard needed
}

__global__ __launch_bounds__(256) void k_bias_init(float* __restrict__ dst,
                                                   const float* __restrict__ b, int total) {
  int i = blockIdx.x * 256 + threadIdx.x;
  if (i < total) dst[i] = b[i & (OUTD - 1)];
}

// Pre-swizzle W (K x 128 f32, row-major) into per-lane contiguous bf16 B-fragments.
// Fragment element i of lane l, tile t, kstep s:  B[K = s*32 + (l>=16?16:0) + i][N = t*16 + (l&15)]
// stored at out[(((s*8 + t)*32 + l)*16 + i]
__global__ __launch_bounds__(256) void k_swizzle_b(const float* __restrict__ W,
                                                   unsigned short* __restrict__ out, int K) {
  int idx = blockIdx.x * 256 + threadIdx.x;
  int total = K * OUTD;
  if (idx >= total) return;
  int i = idx & 15;
  int l = (idx >> 4) & 31;
  int t = (idx >> 9) & 7;
  int s = idx >> 12;
  int krow = s * 32 + ((l & 16) ? 16 : 0) + i;
  int col  = t * 16 + (l & 15);
  out[idx] = f32_to_bf16_bits(W[(size_t)krow * OUTD + col]);
}

// ---------------- WMMA GEMM:  C[nrows x 128] = A[nrows x K] * B ----------------
// Block = 256 threads = 8 waves; block covers 128 rows; wave w -> rows [w*16, w*16+16).
// B (pre-swizzled bf16) staged in LDS via the Tensor Data Mover (one DMA per block,
// completed before the block barrier); A converted f32->bf16 on the fly.

__global__ __launch_bounds__(256) void k_gemm_bf16(const float* __restrict__ A,
                                                   const unsigned short* __restrict__ Bsw,
                                                   float* __restrict__ C,
                                                   int nrows, int K) {
  __shared__ __align__(32) unsigned short sB[32768];   // up to K=256: K*128 bf16 = 64 KB
  const int ksteps = K >> 5;

#if defined(HAVE_TDM)
  if (threadIdx.x < 32) {   // one wave issues the DMA; TDM ignores EXEC
    const unsigned long long ga = (unsigned long long)Bsw;
    // D# group 0: count=1 user descriptor | lds_addr=0 (sB is the only LDS object)
    //             | global_addr bits[120:64] | type=2 ("image") bits[127:126]
    u32x4 g0 = { 1u, 0u,
                 (unsigned)(ga & 0xFFFFFFFFu),
                 (unsigned)((ga >> 32) & 0x01FFFFFFu) | (2u << 30) };
    // 2D tile: rows of 8192 bf16 (16 KB), K/64 rows, contiguous (stride = 8192 elems)
    const unsigned td0  = 8192u;
    const unsigned nrow = (unsigned)(K * OUTD) / td0;   // 4 (K=256) or 2 (K=128)
    i32x8 g1;
    g1[0] = (int)(1u << 16);                 // workgroup_mask=0 | data_size=1 (2 bytes)
    g1[1] = (int)((td0 & 0xFFFFu) << 16);    // tensor_dim0[15:0] in bits 63:48
    g1[2] = (int)((td0 >> 16) | (nrow << 16)); // tensor_dim0[31:16] | tensor_dim1[15:0]
    g1[3] = (int)((td0 & 0xFFFFu) << 16);    // tensor_dim1[31:16]=0 | tile_dim0
    g1[4] = (int)nrow;                       // tile_dim1 | tile_dim2=0
    g1[5] = (int)td0;                        // tensor_dim0_stride[31:0]
    g1[6] = 0;                               // stride0[47:32]=0 | tensor_dim1_stride lo
    g1[7] = 0;
    i32x4 gz = {0, 0, 0, 0};
    TDM_LOAD(g0, g1, gz, gz);
    __builtin_amdgcn_s_wait_tensorcnt(0);
  }
#else
  {
    const int nvec = (K * OUTD) >> 3;                  // uint4 = 8 bf16
    const uint4* src = (const uint4*)Bsw;
    uint4* dst = (uint4*)sB;
    for (int i = threadIdx.x; i < nvec; i += 256) dst[i] = src[i];
  }
#endif
  // The TDM writes LDS through a descriptor, not a pointer LLVM can alias to sB.
  // Escape sB into a may-write asm so the ds_loads below can't be folded to undef.
  asm volatile("" :: "v"((void*)sB) : "memory");
  __syncthreads();

  const int wave = threadIdx.x >> 5;
  const int lane = threadIdx.x & 31;
  const int m    = lane & 15;
  const int lhi  = (lane >> 4) & 1;

  // A-fragment row for this lane (clamped so EXEC stays all-1s for WMMA)
  int arow = blockIdx.x * 128 + wave * 16 + m;
  if (arow > nrows - 1) arow = nrows - 1;
  const float* __restrict__ ap = A + (size_t)arow * K;

  v8f acc[8];
  const v8f vzero = {0.f, 0.f, 0.f, 0.f, 0.f, 0.f, 0.f, 0.f};
#pragma unroll
  for (int t = 0; t < 8; ++t) acc[t] = vzero;

  for (int s = 0; s < ksteps; ++s) {
    const int kb = s * 32;
    // A k order used here (matches B swizzle): lanes 0-15 hold K {0..15}, lanes 16-31 hold K {16..31}
    const float4 a0 = *(const float4*)(ap + kb + lhi * 16);
    const float4 a1 = *(const float4*)(ap + kb + lhi * 16 + 4);
    const float4 a2 = *(const float4*)(ap + kb + lhi * 16 + 8);
    const float4 a3 = *(const float4*)(ap + kb + lhi * 16 + 12);
    bf16x16 af;
    af[0]  = f32_to_bf16(a0.x); af[1]  = f32_to_bf16(a0.y);
    af[2]  = f32_to_bf16(a0.z); af[3]  = f32_to_bf16(a0.w);
    af[4]  = f32_to_bf16(a1.x); af[5]  = f32_to_bf16(a1.y);
    af[6]  = f32_to_bf16(a1.z); af[7]  = f32_to_bf16(a1.w);
    af[8]  = f32_to_bf16(a2.x); af[9]  = f32_to_bf16(a2.y);
    af[10] = f32_to_bf16(a2.z); af[11] = f32_to_bf16(a2.w);
    af[12] = f32_to_bf16(a3.x); af[13] = f32_to_bf16(a3.y);
    af[14] = f32_to_bf16(a3.z); af[15] = f32_to_bf16(a3.w);
#pragma unroll
    for (int t = 0; t < 8; ++t) {
      const bf16x16 bf = *(const bf16x16*)(sB + (((s * 8 + t) * 32 + lane) << 4));
      acc[t] = __builtin_amdgcn_wmma_f32_16x16x32_bf16(
          /*neg_a=*/false, af, /*neg_b=*/false, bf,
          /*c_mod=*/(short)0, acc[t], /*reuse_a=*/false, /*reuse_b=*/false);
    }
  }

  // C layout: element r of acc -> row (lhi?8:0)+r of the wave tile, col = t*16 + (lane&15)
  const int srow0 = blockIdx.x * 128 + wave * 16 + lhi * 8;
#pragma unroll
  for (int t = 0; t < 8; ++t) {
    const int col = t * 16 + m;
#pragma unroll
    for (int r = 0; r < 8; ++r) {
      const int rr = srow0 + r;
      if (rr < nrows) C[(size_t)rr * OUTD + col] = acc[t][r];
    }
  }
}

// ---------------- edge scatter (wave per edge, float4 per lane) ----------------

__global__ __launch_bounds__(256) void k_edge_scatter1(const long long* __restrict__ ei,
                                                       const float* __restrict__ ew,
                                                       const float* __restrict__ dis,
                                                       const float* __restrict__ src,
                                                       float* __restrict__ acc, int E) {
  const int e = blockIdx.x * 8 + (threadIdx.x >> 5);
  if (e >= E) return;
  const int lane = threadIdx.x & 31;
  const int r = (int)ei[e];
  const int c = (int)ei[(size_t)E + e];
  const float nrm = dis[r] * ew[e] * dis[c];
  const float4 v = *(const float4*)(src + (size_t)r * OUTD + lane * 4);
  float* d = acc + (size_t)c * OUTD + lane * 4;
  atomic_add_f32(d + 0, v.x * nrm);
  atomic_add_f32(d + 1, v.y * nrm);
  atomic_add_f32(d + 2, v.z * nrm);
  atomic_add_f32(d + 3, v.w * nrm);
}

__global__ __launch_bounds__(256) void k_edge_scatter2(const long long* __restrict__ ei,
                                                       const float* __restrict__ ew,
                                                       const float* __restrict__ dis,
                                                       const float* __restrict__ smu,
                                                       const float* __restrict__ svar,
                                                       float* __restrict__ omu,
                                                       float* __restrict__ ovar, int E) {
  const int e = blockIdx.x * 8 + (threadIdx.x >> 5);
  if (e >= E) return;
  const int lane = threadIdx.x & 31;
  const int r = (int)ei[e];
  const int c = (int)ei[(size_t)E + e];
  const float nrm = dis[r] * ew[e] * dis[c];
  const size_t ro = (size_t)r * OUTD + lane * 4;
  const size_t co = (size_t)c * OUTD + lane * 4;
  const float4 vm = *(const float4*)(smu + ro);
  const float4 vv = *(const float4*)(svar + ro);
  float* dm = omu + co;
  float* dv = ovar + co;
  atomic_add_f32(dm + 0, vm.x * nrm); atomic_add_f32(dm + 1, vm.y * nrm);
  atomic_add_f32(dm + 2, vm.z * nrm); atomic_add_f32(dm + 3, vm.w * nrm);
  atomic_add_f32(dv + 0, vv.x * nrm); atomic_add_f32(dv + 1, vv.y * nrm);
  atomic_add_f32(dv + 2, vv.z * nrm); atomic_add_f32(dv + 3, vv.w * nrm);
}

// self-loop contribution (norm = dis[i]*2*dis[i]) + ReLU for layer 1
__global__ __launch_bounds__(256) void k_self_relu(const float* __restrict__ xw,
                                                   const float* __restrict__ dis,
                                                   float* __restrict__ feat, int total) {
  int i = blockIdx.x * 256 + threadIdx.x;
  if (i >= total) return;
  const int node = i >> 7;
  const float d = dis[node];
  const float sn = 2.0f * d * d;
  const float v = feat[i] + xw[i] * sn;
  feat[i] = v > 0.0f ? v : 0.0f;
}

// self-loop contribution for mu and var outputs
__global__ __launch_bounds__(256) void k_self_add2(const float* __restrict__ fmu,
                                                   const float* __restrict__ fvar,
                                                   const float* __restrict__ dis,
                                                   float* __restrict__ omu,
                                                   float* __restrict__ ovar, int total) {
  int i = blockIdx.x * 256 + threadIdx.x;
  if (i >= total) return;
  const int node = i >> 7;
  const float d = dis[node];
  const float sn = 2.0f * d * d;
  omu[i]  += fmu[i] * sn;
  ovar[i] += fvar[i] * sn;
}

// ---------------- launcher ----------------

extern "C" void kernel_launch(void* const* d_in, const int* in_sizes, int n_in,
                              void* d_out, int out_size, void* d_ws, size_t ws_size,
                              hipStream_t stream) {
  const float*     x    = (const float*)d_in[0];
  const long long* ei   = (const long long*)d_in[1];   // int64 edge_index (2, E) flat
  const float*     ew   = (const float*)d_in[2];
  const float*     W1   = (const float*)d_in[3];
  const float*     b1   = (const float*)d_in[4];
  const float*     Wmu  = (const float*)d_in[5];
  const float*     bmu  = (const float*)d_in[6];
  const float*     Wvar = (const float*)d_in[7];
  const float*     bvar = (const float*)d_in[8];

  const int N = in_sizes[0] / 256;   // IN_DIM = 256
  const int E = in_sizes[2];

  float* out_mu  = (float*)d_out;
  float* out_var = out_mu + (size_t)N * OUTD;

  // workspace layout (floats)
  float* ws   = (float*)d_ws;
  const size_t n128 = (size_t)N * OUTD;
  float* dis  = ws;                          // N floats (deg, then rsqrt in place)
  float* xw   = ws + ((N + 3) & ~3);         // N x 128
  float* feat = xw + n128;                   // N x 128 (accumulator, then relu'd features)
  float* fmu  = feat + n128;                 // N x 128
  float* fvar = fmu + n128;                  // N x 128
  unsigned short* bsw1   = (unsigned short*)(fvar + n128);  // 256*128 bf16
  unsigned short* bswmu  = bsw1 + 256 * OUTD;               // 128*128 bf16
  unsigned short* bswvar = bswmu + OUTD * OUTD;             // 128*128 bf16

  const int total = (int)n128;
  const dim3 blk(256);

  // 1) degrees -> dis = rsqrt(deg), deg starts at self-loop weight 2.0
  k_fill<<<(N + 255) / 256, blk, 0, stream>>>(dis, 2.0f, N);
  k_deg_accum<<<(E + 255) / 256, blk, 0, stream>>>(ei, ew, dis, E);
  k_rsqrt<<<(N + 255) / 256, blk, 0, stream>>>(dis, N);

  // 2) swizzle weights into WMMA B-fragment layout (bf16)
  k_swizzle_b<<<(256 * OUTD) / 256, blk, 0, stream>>>(W1, bsw1, 256);
  k_swizzle_b<<<(OUTD * OUTD) / 256, blk, 0, stream>>>(Wmu, bswmu, OUTD);
  k_swizzle_b<<<(OUTD * OUTD) / 256, blk, 0, stream>>>(Wvar, bswvar, OUTD);

  // 3) layer 1: xw = x @ W1 (WMMA), scatter, self-loop + bias + relu
  k_gemm_bf16<<<(N + 127) / 128, blk, 0, stream>>>(x, bsw1, xw, N, 256);
  k_bias_init<<<(total + 255) / 256, blk, 0, stream>>>(feat, b1, total);
  k_edge_scatter1<<<(E + 7) / 8, blk, 0, stream>>>(ei, ew, dis, xw, feat, E);
  k_self_relu<<<(total + 255) / 256, blk, 0, stream>>>(xw, dis, feat, total);

  // 4) layer 2: mu and var heads
  k_gemm_bf16<<<(N + 127) / 128, blk, 0, stream>>>(feat, bswmu, fmu, N, OUTD);
  k_gemm_bf16<<<(N + 127) / 128, blk, 0, stream>>>(feat, bswvar, fvar, N, OUTD);
  k_bias_init<<<(total + 255) / 256, blk, 0, stream>>>(out_mu, bmu, total);
  k_bias_init<<<(total + 255) / 256, blk, 0, stream>>>(out_var, bvar, total);
  k_edge_scatter2<<<(E + 7) / 8, blk, 0, stream>>>(ei, ew, dis, fmu, fvar, out_mu, out_var, E);
  k_self_add2<<<(total + 255) / 256, blk, 0, stream>>>(fmu, fvar, dis, out_mu, out_var, total);
}
